// LatentExpander_27135603376314
// MI455X (gfx1250) — matrix operand order repaired and checked by reading, not compiled, not run on hardware
//
#include <hip/hip_runtime.h>
#include <hip/hip_bf16.h>

typedef __attribute__((ext_vector_type(16))) _Float16 v16h;
typedef __attribute__((ext_vector_type(8)))  _Float16 v8h;
typedef __attribute__((ext_vector_type(8)))  float    v8f;

#define H_DIM 128
#define SD    256
#define LD    64
#define G4    512          // 4*H gates
#define WGB   32           // batch rows per workgroup
#define NBLK  16           // 512 / WGB
#define K0    192          // dynamic K for layer0: [h_prev(64) | h_l0(128)]
#define K1    256          // K for layer1: [h_l0(128) | h_l1(128)]

// ---------------- workspace layout (bytes) ----------------
// Weights stored pre-swizzled in WMMA B-fragment order:
//   fragment (kt, nt) = 512 halves: [ lane*8 lo halves | +256: lane*8 hi halves ]
// Gz stored in WMMA D-fragment order:
//   fragment = 256 floats: lane*8 + r
#define WS_W0T   0                          // 192 frags * 1KB  (layer0: 6 kt x 32 nt)
#define WS_W1T   (WS_W0T + 192*512*2)       // 256 frags * 1KB  (layer1: 8 kt x 32 nt)
#define WS_WOT   (WS_W1T + 256*512*2)       // 16 frags * 1KB   (out: 4 kt x 4 nt)
#define WS_GZ    (WS_WOT + 128*64*2)        // 1024 frags * 1KB (16 bt x 2 mt x 32 nt)
#define WS_H0    (WS_GZ  + 512*512*4)       // 512x128 f16
#define WS_H1    (WS_H0  + 512*128*2)       // 512x128 f16
#define WS_C0    (WS_H1  + 512*128*2)       // 512x128 f32
#define WS_C1    (WS_C0  + 512*128*4)       // 512x128 f32

// ---------------- WMMA helpers (CDNA5 16x16x32 f16 -> f32) ----------------
__device__ __forceinline__ v8f wmma16(v16h a, v16h b, v8f c) {
  return __builtin_amdgcn_wmma_f32_16x16x32_f16(false, a, false, b, (short)0, c, false, false);
}

__device__ __forceinline__ v16h cat8(v8h lo, v8h hi) {
  return __builtin_shufflevector(lo, hi, 0,1,2,3,4,5,6,7,8,9,10,11,12,13,14,15);
}

// A-operand (16x32 f16) from LDS, row-major [rows][pitch].
// ISA layout: lane M = lane&15; lanes>=16 shifted +8 in K; elems 0..7 -> K+0..7, elems 8..15 -> K+16..23
__device__ __forceinline__ v16h load_a_lds(const _Float16* A, int pitch, int row0, int kb, int lane) {
  const _Float16* p = A + (size_t)(row0 + (lane & 15)) * pitch + kb + ((lane & 16) ? 8 : 0);
  v8h lo = *(const v8h*)(p);
  v8h hi = *(const v8h*)(p + 16);
  return cat8(lo, hi);
}

// B-operand from pre-swizzled global weights: fully coalesced (2 x 512B per wave).
__device__ __forceinline__ v16h load_b_sw(const _Float16* Wsw, int frag, int lane) {
  const _Float16* p = Wsw + (size_t)frag * 512 + lane * 8;
  v8h lo = *(const v8h*)(p);
  v8h hi = *(const v8h*)(p + 256);
  return cat8(lo, hi);
}

__device__ __forceinline__ float fsig(float x) {
  x = fminf(20.f, fmaxf(-20.f, x));
  return __builtin_amdgcn_rcpf(1.f + __expf(-x));
}
__device__ __forceinline__ float ftanh(float x) {
  x = fminf(15.f, fmaxf(-15.f, x));
  float e = __expf(-2.f * x);
  return (1.f - e) * __builtin_amdgcn_rcpf(1.f + e);
}

// ---------------- prep kernels ----------------
// swizzled fragment offset for a K-major weight element (k, n), NT = N/16 tiles
__device__ __forceinline__ size_t bswz(int k, int n, int NT) {
  int kt = k >> 5, kk = k & 31, nt = n >> 4, lane = ((kk & 8) << 1) | (n & 15);
  return (size_t)(kt * NT + nt) * 512 + ((kk & 16) ? 256 : 0) + lane * 8 + (kk & 7);
}

__global__ void prep_weights(const float* __restrict__ Wih0, const float* __restrict__ Whh0,
                             const float* __restrict__ Wih1, const float* __restrict__ Whh1,
                             const float* __restrict__ Wout,
                             _Float16* __restrict__ w0t, _Float16* __restrict__ w1t,
                             _Float16* __restrict__ wot) {
  int idx = blockIdx.x * blockDim.x + threadIdx.x;
  if (idx < 192 * 512) {
    int k = idx >> 9, n = idx & 511;
    float v = (k < 64) ? Wih0[n * 320 + k] : Whh0[n * 128 + (k - 64)];
    w0t[bswz(k, n, 32)] = (_Float16)v;
  } else if (idx < 192 * 512 + 256 * 512) {
    int i = idx - 192 * 512;
    int k = i >> 9, n = i & 511;
    float v = (k < 128) ? Wih1[n * 128 + k] : Whh1[n * 128 + (k - 128)];
    w1t[bswz(k, n, 32)] = (_Float16)v;
  } else if (idx < 192 * 512 + 256 * 512 + 128 * 64) {
    int i = idx - (192 * 512 + 256 * 512);
    int k = i >> 6, n = i & 63;
    wot[bswz(k, n, 4)] = (_Float16)Wout[n * 128 + k];
  }
}

__global__ void prep_gz(const float* __restrict__ z, const float* __restrict__ Wih0,
                        const float* __restrict__ bih0, const float* __restrict__ bhh0,
                        float* __restrict__ gz) {
  int idx = blockIdx.x * blockDim.x + threadIdx.x;   // 512*512
  int b = idx >> 9, n = idx & 511;
  const float* zp = z + (size_t)b * SD;
  const float* wp = Wih0 + (size_t)n * 320 + 64;     // z occupies cols 64..319 of W_ih0
  float s = bih0[n] + bhh0[n];
  for (int i = 0; i < SD; ++i) s += zp[i] * wp[i];
  // store in D-fragment order
  int bt = b >> 5, mrem = b & 31, mt = mrem >> 4, m16 = mrem & 15;
  int lane = ((m16 >> 3) << 4) | (n & 15);
  int r = m16 & 7;
  int frag = (bt * 2 + mt) * 32 + (n >> 4);
  gz[(size_t)frag * 256 + lane * 8 + r] = s;
}

__global__ void prep_init(const float* __restrict__ z,
                          const float* __restrict__ Wih, const float* __restrict__ bh,
                          const float* __restrict__ Wic, const float* __restrict__ bc,
                          _Float16* __restrict__ h0, _Float16* __restrict__ h1,
                          float* __restrict__ c0, float* __restrict__ c1) {
  int idx = blockIdx.x * blockDim.x + threadIdx.x;   // 512*256
  int b = idx >> 8, n = idx & 255;
  const float* zp = z + (size_t)b * SD;
  const float* wh = Wih + (size_t)n * SD;
  const float* wc = Wic + (size_t)n * SD;
  float sh = bh[n], sc = bc[n];
  for (int i = 0; i < SD; ++i) { sh += zp[i] * wh[i]; sc += zp[i] * wc[i]; }
  if (n < 128) { h0[b * 128 + n] = (_Float16)sh; c0[b * 128 + n] = sc; }
  else         { h1[b * 128 + n - 128] = (_Float16)sh; c1[b * 128 + n - 128] = sc; }
}

// ---------------- persistent recurrence kernel ----------------
// 16 blocks x 512 threads. Each block owns 32 batch rows.
// Wave wv: mt = wv>>3 (m-tile 0..1), g = wv&7 (n-group / j-tile 0..7).
// Wave's 4 gate n-tiles {g, g+8, g+16, g+24} == the i/f/g/o quadruple for j-tile g,
// so the cell update is wave-local and cell state c lives in VGPRs all run long.
// Weights stream (coalesced, fragment-order) from L2 every step; an opaque integer
// offset stops LICM from hoisting-and-spilling them while keeping GLOBAL addressing.
__global__ __launch_bounds__(512) void lstm_persist(
    const float* __restrict__ gz_in,
    const _Float16* __restrict__ w0t_in, const _Float16* __restrict__ w1t_in,
    const _Float16* __restrict__ wot_in,
    const _Float16* __restrict__ h0i, const _Float16* __restrict__ h1i,
    const float* __restrict__ c0i, const float* __restrict__ c1i,
    const float* __restrict__ bih1, const float* __restrict__ bhh1,
    const float* __restrict__ bout,
    const int* __restrict__ seqp,
    float* __restrict__ out) {
  __shared__ _Float16 A0[WGB][K0];   // [h_prev(64) | h_l0(128)]
  __shared__ _Float16 A1[WGB][K1];   // [h_l0(128)  | h_l1(128)]

  const int tid  = threadIdx.x;
  const int lane = tid & 31;
  const int wv   = tid >> 5;          // 0..15
  const int mt   = wv >> 3;           // 0..1
  const int g    = wv & 7;            // 0..7
  const int b0   = blockIdx.x * WGB;
  const int T    = seqp[0];
  const int jlo  = lane & 15;
  const int mhi  = (lane & 16) ? 8 : 0;

  // ---- init LDS state ----
  for (int i = tid; i < WGB * LD; i += 512)
    A0[i / LD][i % LD] = (_Float16)0.f;                 // h_prev0 = 0
  for (int i = tid; i < WGB * H_DIM; i += 512) {
    int m = i / H_DIM, j = i % H_DIM;
    _Float16 v0 = h0i[(size_t)(b0 + m) * H_DIM + j];
    A0[m][LD + j]    = v0;
    A1[m][j]         = v0;
    A1[m][H_DIM + j] = h1i[(size_t)(b0 + m) * H_DIM + j];
  }

  // ---- persistent per-wave registers ----
  float b1v[4];
#pragma unroll
  for (int p = 0; p < 4; ++p) {
    int n = (g + 8 * p) * 16 + jlo;
    b1v[p] = bih1[n] + bhh1[n];
  }
  const bool owave = (g < 4);                           // out-projection waves
  float bov = bout[(g & 3) * 16 + jlo];

  v8f c0f, c1f;                                         // f32 cell state in VGPRs
#pragma unroll
  for (int r = 0; r < 8; ++r) {
    size_t idx = (size_t)(b0 + mt * 16 + mhi + r) * H_DIM + g * 16 + jlo;
    c0f[r] = c0i[idx];
    c1f[r] = c1i[idx];
  }
  __syncthreads();

  const int gzfrag0 = (blockIdx.x * 2 + mt) * 32;       // this wave's Gz fragment base

  // Opaque, loop-"variant" zero offset: blocks LICM of the weight loads while
  // keeping the base pointers recognizable as GLOBAL for addrspace inference.
  int zoff = 0;

  for (int t = 0; t < T; ++t) {
    asm volatile("" : "+s"(zoff));
    const _Float16* w0p = w0t_in + zoff;
    const _Float16* w1p = w1t_in + zoff;
    const _Float16* wop = wot_in + zoff;
    const float*    gzp = gz_in  + zoff;

    // ========== layer 0: gates = [h_prev|h_l0] @ Wcat0 + Gz ==========
    v8f C[4];
#pragma unroll
    for (int p = 0; p < 4; ++p)
      C[p] = *(const v8f*)(gzp + (size_t)(gzfrag0 + g + 8 * p) * 256 + lane * 8);

    for (int kt = 0; kt < K0 / 32; ++kt) {
      v16h a = load_a_lds(&A0[0][0], K0, mt * 16, kt * 32, lane);
#pragma unroll
      for (int p = 0; p < 4; ++p) {
        v16h b = load_b_sw(w0p, kt * 32 + g + 8 * p, lane);
        C[p] = wmma16(a, b, C[p]);
      }
    }
    __syncthreads();                                   // all A0 reads done
#pragma unroll
    for (int r = 0; r < 8; ++r) {                      // C[0]=i, C[1]=f, C[2]=g, C[3]=o  (j-tile g)
      float iv = fsig (C[0][r]);
      float fv = fsig (C[1][r]);
      float gv = ftanh(C[2][r]);
      float ov = fsig (C[3][r]);
      float cn = fv * c0f[r] + iv * gv;
      c0f[r] = cn;
      float hn = ov * ftanh(cn);
      int m = mt * 16 + mhi + r;
      int j = g * 16 + jlo;
      _Float16 hh = (_Float16)hn;
      A0[m][LD + j] = hh;
      A1[m][j]      = hh;
    }
    __syncthreads();                                   // h_l0 visible

    // ========== layer 1: gates = [h_l0|h_l1] @ Wcat1 + b ==========
#pragma unroll
    for (int p = 0; p < 4; ++p)
#pragma unroll
      for (int r = 0; r < 8; ++r) C[p][r] = b1v[p];

    for (int kt = 0; kt < K1 / 32; ++kt) {
      v16h a = load_a_lds(&A1[0][0], K1, mt * 16, kt * 32, lane);
#pragma unroll
      for (int p = 0; p < 4; ++p) {
        v16h b = load_b_sw(w1p, kt * 32 + g + 8 * p, lane);
        C[p] = wmma16(a, b, C[p]);
      }
    }
    __syncthreads();                                   // all A1 reads done
#pragma unroll
    for (int r = 0; r < 8; ++r) {
      float iv = fsig (C[0][r]);
      float fv = fsig (C[1][r]);
      float gv = ftanh(C[2][r]);
      float ov = fsig (C[3][r]);
      float cn = fv * c1f[r] + iv * gv;
      c1f[r] = cn;
      float hn = ov * ftanh(cn);
      A1[mt * 16 + mhi + r][H_DIM + g * 16 + jlo] = (_Float16)hn;
    }
    __syncthreads();                                   // h_l1 visible

    // ========== output: h_t = tanh(h_l1 @ W_out.T + b_out) ==========
    if (owave) {                                       // wave-uniform branch, EXEC stays full
      v8f Co;
#pragma unroll
      for (int r = 0; r < 8; ++r) Co[r] = bov;
      for (int kt = 0; kt < 4; ++kt) {
        v16h a = load_a_lds(&A1[0][0], K1, mt * 16, H_DIM + kt * 32, lane);
        v16h b = load_b_sw(wop, kt * 4 + (g & 3), lane);
        Co = wmma16(a, b, Co);
      }
#pragma unroll
      for (int r = 0; r < 8; ++r) {
        float hv = ftanh(Co[r]);
        int m = mt * 16 + mhi + r;
        int j = (g & 3) * 16 + jlo;
        A0[m][j] = (_Float16)hv;                       // feeds next step's layer0
        out[(size_t)(b0 + m) * T * LD + (size_t)t * LD + j] = hv;
      }
    }
    __syncthreads();
  }
}

// ---------------- host ----------------
extern "C" void kernel_launch(void* const* d_in, const int* in_sizes, int n_in,
                              void* d_out, int out_size, void* d_ws, size_t ws_size,
                              hipStream_t stream) {
  const float* z        = (const float*)d_in[0];
  const int*   seqlen   = (const int*)  d_in[1];
  const float* W_init_h = (const float*)d_in[2];
  const float* b_init_h = (const float*)d_in[3];
  const float* W_init_c = (const float*)d_in[4];
  const float* b_init_c = (const float*)d_in[5];
  const float* W_ih0    = (const float*)d_in[6];
  const float* W_hh0    = (const float*)d_in[7];
  const float* b_ih0    = (const float*)d_in[8];
  const float* b_hh0    = (const float*)d_in[9];
  const float* W_ih1    = (const float*)d_in[10];
  const float* W_hh1    = (const float*)d_in[11];
  const float* b_ih1    = (const float*)d_in[12];
  const float* b_hh1    = (const float*)d_in[13];
  const float* W_out    = (const float*)d_in[14];
  const float* b_out    = (const float*)d_in[15];

  char* ws = (char*)d_ws;
  _Float16* w0t = (_Float16*)(ws + WS_W0T);
  _Float16* w1t = (_Float16*)(ws + WS_W1T);
  _Float16* wot = (_Float16*)(ws + WS_WOT);
  float*    gz  = (float*)   (ws + WS_GZ);
  _Float16* h0  = (_Float16*)(ws + WS_H0);
  _Float16* h1  = (_Float16*)(ws + WS_H1);
  float*    c0  = (float*)   (ws + WS_C0);
  float*    c1  = (float*)   (ws + WS_C1);

  prep_weights<<<(192 * 512 + 256 * 512 + 128 * 64 + 255) / 256, 256, 0, stream>>>(
      W_ih0, W_hh0, W_ih1, W_hh1, W_out, w0t, w1t, wot);
  prep_gz<<<(512 * 512) / 256, 256, 0, stream>>>(z, W_ih0, b_ih0, b_hh0, gz);
  prep_init<<<(512 * 256) / 256, 256, 0, stream>>>(
      z, W_init_h, b_init_h, W_init_c, b_init_c, h0, h1, c0, c1);

  lstm_persist<<<NBLK, 512, 0, stream>>>(
      gz, w0t, w1t, wot, h0, h1, c0, c1, b_ih1, b_hh1, b_out, seqlen, (float*)d_out);
}
